// SwinTransformerLayer_53128745451953
// MI455X (gfx1250) — matrix operand order repaired and checked by reading, not compiled, not run on hardware
//
#include <hip/hip_runtime.h>
#include <hip/hip_bf16.h>
#include <math.h>

typedef _Float16 v16h __attribute__((ext_vector_type(16)));
typedef _Float16 v8h  __attribute__((ext_vector_type(8)));
typedef float    v8f  __attribute__((ext_vector_type(8)));
typedef int      v4i  __attribute__((ext_vector_type(4)));

#define DIMC   384
#define HEADS  12
#define TOKENS 100352   // 32*56*56

#if defined(__has_builtin)
#if __has_builtin(__builtin_amdgcn_global_load_async_to_lds_b128) && \
    __has_builtin(__builtin_amdgcn_s_wait_asynccnt)
#define USE_ASYNC_LDS 1
#endif
#endif
#ifndef USE_ASYNC_LDS
#define USE_ASYNC_LDS 0
#endif

#if USE_ASYNC_LDS
typedef __attribute__((address_space(1))) v4i* as1_v4i_ptr;
typedef __attribute__((address_space(3))) v4i* as3_v4i_ptr;
__device__ __forceinline__ void async_b128(const _Float16* g, _Float16* l) {
  __builtin_amdgcn_global_load_async_to_lds_b128(
      (as1_v4i_ptr)(g), (as3_v4i_ptr)(l), 0, 0);
}
#endif

// ---------------- WMMA fragment helpers (CDNA5 16x16x32 f16 layouts) ----------------
// A (16xK tile, row-major LDS [m][k]): lane m=l&15, hi=l>>4; K chunks at hi*8 and 16+hi*8
__device__ __forceinline__ v16h ld_fragA(const _Float16* base, int stride, int colOff) {
  int lane = threadIdx.x & 31;
  int m = lane & 15, hi = lane >> 4;
  const _Float16* p = base + m * stride + colOff + hi * 8;
  v8h lo = *(const v8h*)(p);
  v8h hv = *(const v8h*)(p + 16);
  v16h r;
#pragma unroll
  for (int i = 0; i < 8; ++i) { r[i] = lo[i]; r[8 + i] = hv[i]; }
  return r;
}
// B (Kx16 tile stored transposed in LDS as [n][k]): lane n=l&15, hi=l>>4; K chunks at hi*8 and 16+hi*8
__device__ __forceinline__ v16h ld_fragB(const _Float16* base, int stride, int colOff) {
  int lane = threadIdx.x & 31;
  int n = lane & 15, hi = lane >> 4;
  const _Float16* p = base + n * stride + colOff + hi * 8;
  v8h lo = *(const v8h*)(p);
  v8h hv = *(const v8h*)(p + 16);
  v16h r;
#pragma unroll
  for (int i = 0; i < 8; ++i) { r[i] = lo[i]; r[8 + i] = hv[i]; }
  return r;
}
__device__ __forceinline__ v8f wmma_f16(v16h a, v16h b, v8f c) {
  return __builtin_amdgcn_wmma_f32_16x16x32_f16(false, a, false, b, (short)0, c, false, false);
}

// ---------------- fp32 -> f16 convert + transpose (weights: W[K][N] -> Wt[N][K]) -----
__global__ void cvt_transpose_f16(const float* __restrict__ s, _Float16* __restrict__ d,
                                  int K, int N) {
  int i = blockIdx.x * blockDim.x + threadIdx.x;
  if (i < K * N) {
    int k = i / N, n = i % N;
    d[(size_t)n * K + k] = (_Float16)s[i];
  }
}

// ---------------- LayerNorm (one wave32 per token, C=384) ----------------
__global__ __launch_bounds__(256) void ln_f16_kernel(const float* __restrict__ x,
    const float* __restrict__ sc, const float* __restrict__ bi,
    _Float16* __restrict__ out, int tokens) {
  int w = blockIdx.x * 8 + (threadIdx.x >> 5);
  if (w >= tokens) return;
  int lane = threadIdx.x & 31;
  const float* px = x + (size_t)w * DIMC;
  float v[12];
  float s = 0.f;
#pragma unroll
  for (int j = 0; j < 12; ++j) { v[j] = px[lane + j * 32]; s += v[j]; }
#pragma unroll
  for (int m = 1; m < 32; m <<= 1) s += __shfl_xor(s, m, 32);
  float mu = s * (1.0f / DIMC);
  float q = 0.f;
#pragma unroll
  for (int j = 0; j < 12; ++j) { float d = v[j] - mu; q += d * d; }
#pragma unroll
  for (int m = 1; m < 32; m <<= 1) q += __shfl_xor(q, m, 32);
  float rstd = rsqrtf(q * (1.0f / DIMC) + 1e-5f);
  _Float16* po = out + (size_t)w * DIMC;
#pragma unroll
  for (int j = 0; j < 12; ++j) {
    int c = lane + j * 32;
    po[c] = (_Float16)((v[j] - mu) * rstd * sc[c] + bi[c]);
  }
}

// ---- Tiled f16 GEMM, double-buffered LDS: out(MxN) = A(MxK) @ Wt(NxK)^T + bias ------
// EPI 0: store f16;  EPI 1: fp32 out = acc + bias + resid;  EPI 2: f16 out = gelu(acc+bias)
template <int EPI>
__global__ __launch_bounds__(256) void gemm_f16_kernel(
    const _Float16* __restrict__ A, const _Float16* __restrict__ Wt,
    const float* __restrict__ bias, const float* __restrict__ resid,
    void* __restrict__ out, int M, int N, int K) {
  __shared__ __align__(16) _Float16 As[2][128 * 40];
  __shared__ __align__(16) _Float16 Bs[2][128 * 40];
  int tid = threadIdx.x;
  int rowBase = blockIdx.y * 128;
  int colBase = blockIdx.x * 128;
  int wave  = tid >> 5;
  int waveM = (wave >> 2) * 64;   // 2 wave rows
  int waveN = (wave & 3) * 32;    // 4 wave cols
  const v8f vzero = {0.f, 0.f, 0.f, 0.f, 0.f, 0.f, 0.f, 0.f};
  v8f acc[4][2];
#pragma unroll
  for (int i = 0; i < 4; ++i) { acc[i][0] = vzero; acc[i][1] = vzero; }

  // staging map: each thread moves 2x b128 per operand per K-step
  int srow  = tid >> 2;          // 0..63 (also covers srow+64)
  int spart = (tid & 3) * 8;     // 0,8,16,24
  const _Float16* Ag = A  + (size_t)(rowBase + srow) * K + spart;
  const _Float16* Bg = Wt + (size_t)(colBase + srow) * K + spart;
  const size_t gStride = (size_t)64 * K;
  int lsOff0 = srow * 40 + spart;
  int lsOff1 = (srow + 64) * 40 + spart;

#if USE_ASYNC_LDS
  // async tile 0
  async_b128(Ag,           &As[0][lsOff0]);
  async_b128(Ag + gStride, &As[0][lsOff1]);
  async_b128(Bg,           &Bs[0][lsOff0]);
  async_b128(Bg + gStride, &Bs[0][lsOff1]);
  __builtin_amdgcn_s_wait_asynccnt(0);
#else
  v8h pa0 = *(const v8h*)(Ag);
  v8h pa1 = *(const v8h*)(Ag + gStride);
  v8h pb0 = *(const v8h*)(Bg);
  v8h pb1 = *(const v8h*)(Bg + gStride);
  *(v8h*)(&As[0][lsOff0]) = pa0;
  *(v8h*)(&As[0][lsOff1]) = pa1;
  *(v8h*)(&Bs[0][lsOff0]) = pb0;
  *(v8h*)(&Bs[0][lsOff1]) = pb1;
#endif
  __syncthreads();

  int nIter = K >> 5;
  for (int it = 0; it < nIter; ++it) {
    int cur = it & 1, nxt = cur ^ 1;
    bool hasNext = (it + 1) < nIter;
    if (hasNext) {                      // prefetch next K-slab while computing
      const _Float16* ag = Ag + (it + 1) * 32;
      const _Float16* bg = Bg + (it + 1) * 32;
#if USE_ASYNC_LDS
      async_b128(ag,           &As[nxt][lsOff0]);
      async_b128(ag + gStride, &As[nxt][lsOff1]);
      async_b128(bg,           &Bs[nxt][lsOff0]);
      async_b128(bg + gStride, &Bs[nxt][lsOff1]);
#else
      pa0 = *(const v8h*)(ag);
      pa1 = *(const v8h*)(ag + gStride);
      pb0 = *(const v8h*)(bg);
      pb1 = *(const v8h*)(bg + gStride);
#endif
    }
    v16h af[4], bf[2];
#pragma unroll
    for (int mt = 0; mt < 4; ++mt) af[mt] = ld_fragA(&As[cur][(waveM + mt * 16) * 40], 40, 0);
#pragma unroll
    for (int nt = 0; nt < 2; ++nt) bf[nt] = ld_fragB(&Bs[cur][(waveN + nt * 16) * 40], 40, 0);
#pragma unroll
    for (int mt = 0; mt < 4; ++mt)
#pragma unroll
      for (int nt = 0; nt < 2; ++nt)
        acc[mt][nt] = wmma_f16(af[mt], bf[nt], acc[mt][nt]);
    if (hasNext) {
#if USE_ASYNC_LDS
      __builtin_amdgcn_s_wait_asynccnt(0);
#else
      *(v8h*)(&As[nxt][lsOff0]) = pa0;
      *(v8h*)(&As[nxt][lsOff1]) = pa1;
      *(v8h*)(&Bs[nxt][lsOff0]) = pb0;
      *(v8h*)(&Bs[nxt][lsOff1]) = pb1;
#endif
    }
    __syncthreads();
  }

  int lane = tid & 31;
  int nl = lane & 15, hi = lane >> 4;
#pragma unroll
  for (int nt = 0; nt < 2; ++nt) {
    int col = colBase + waveN + nt * 16 + nl;
    float bv = bias[col];
#pragma unroll
    for (int mt = 0; mt < 4; ++mt) {
#pragma unroll
      for (int r = 0; r < 8; ++r) {
        int row = rowBase + waveM + mt * 16 + r + 8 * hi;
        size_t idx = (size_t)row * N + col;
        float v = acc[mt][nt][r] + bv;
        if (EPI == 0) {
          ((_Float16*)out)[idx] = (_Float16)v;
        } else if (EPI == 1) {
          ((float*)out)[idx] = v + resid[idx];
        } else {
          ((_Float16*)out)[idx] =
              (_Float16)(0.5f * v * (1.0f + erff(v * 0.70710678118654752f)));
        }
      }
    }
  }
}

// ---------------- Shifted-window attention: one block per (window, head) ------------
__global__ __launch_bounds__(128) void attn_kernel(
    const _Float16* __restrict__ qkv, const float* __restrict__ rpb,
    _Float16* __restrict__ outp) {
  __shared__ __align__(16) _Float16 Qs[64 * 40];
  __shared__ __align__(16) _Float16 Ks[64 * 40];
  __shared__ __align__(16) _Float16 Vt[32 * 72];
  __shared__ __align__(16) _Float16 Ps[4][16 * 72];
  __shared__ int   tok[64];
  __shared__ float bcol[169];

  int head = blockIdx.x;
  int win  = blockIdx.y;
  int tid  = threadIdx.x;
  int b  = win >> 6;
  int wi = win & 63;
  int wh = wi >> 3;
  int ww = wi & 7;

  if (tid < 64) {
    int p = tid;
    int i = p / 7, j = p % 7;
    int hs = wh * 7 + i, wsc = ww * 7 + j;
    int h = (hs + 3) % 56, w2 = (wsc + 3) % 56;     // inverse of roll(-shift)
    tok[p] = (p < 49) ? (b * 3136 + h * 56 + w2) : 0;
  }
  for (int idx = tid; idx < 169; idx += 128) bcol[idx] = rpb[idx * HEADS + head];
  __syncthreads();

  { // gather Q,K into [pos][d]; V transposed into [d][pos]; pad pos>=49 with zeros
    int d = tid & 31;
    for (int p = tid >> 5; p < 64; p += 4) {
      _Float16 qv = (_Float16)0.f, kv = (_Float16)0.f, vv = (_Float16)0.f;
      if (p < 49) {
        size_t base = (size_t)tok[p] * 1152 + head * 32 + d;
        qv = qkv[base]; kv = qkv[base + 384]; vv = qkv[base + 768];
      }
      Qs[p * 40 + d] = qv;
      Ks[p * 40 + d] = kv;
      Vt[d * 72 + p] = vv;
    }
  }
  __syncthreads();

  int mt   = tid >> 5;       // query strip (0..3)
  int lane = tid & 31;
  int nl = lane & 15, hi = lane >> 4;

  const v8f vzero = {0.f, 0.f, 0.f, 0.f, 0.f, 0.f, 0.f, 0.f};
  v8f sc[4] = {vzero, vzero, vzero, vzero};
  v16h aq = ld_fragA(Qs + mt * 16 * 40, 40, 0);
#pragma unroll
  for (int j2 = 0; j2 < 4; ++j2) {
    v16h bk = ld_fragB(Ks + j2 * 16 * 40, 40, 0);
    sc[j2] = wmma_f16(aq, bk, sc[j2]);
  }

  const float scale = 0.17677669529663687f;  // 1/sqrt(32)
  bool edgeH = (wh == 7), edgeW = (ww == 7);

  // key-side quantities depend only on (j2, lane): hoist out of the row loop
  bool kvalid[4];
  int  kidx[4], regk[4];
#pragma unroll
  for (int j2 = 0; j2 < 4; ++j2) {
    int kp = j2 * 16 + nl;
    kvalid[j2] = (kp < 49);
    int kpc = kvalid[j2] ? kp : 48;
    int kr = kpc / 7, kc = kpc % 7;
    kidx[j2] = kr * 13 + kc;
    regk[j2] = (edgeH ? (kr < 4 ? 1 : 2) : 0) * 3 + (edgeW ? (kc < 4 ? 1 : 2) : 0);
  }

#pragma unroll
  for (int r = 0; r < 8; ++r) {
    int qp = mt * 16 + r + 8 * hi;            // query position (garbage rows >=49 ignored)
    int qi = qp < 49 ? qp : 48;
    int qr = qi / 7, qc = qi % 7;
    int qidx = qr * 13 + qc;
    int regq = (edgeH ? (qr < 4 ? 1 : 2) : 0) * 3 + (edgeW ? (qc < 4 ? 1 : 2) : 0);
    float lv[4];
    float mx = -3.0e38f;
#pragma unroll
    for (int j2 = 0; j2 < 4; ++j2) {
      float lg = -3.0e38f;
      if (kvalid[j2]) {
        float bias = bcol[qidx - kidx[j2] + 84];
        float msk  = (regq == regk[j2]) ? 0.f : -100.f;
        lg = sc[j2][r] * scale + bias + msk;
      }
      lv[j2] = lg;
      mx = fmaxf(mx, lg);
    }
#pragma unroll
    for (int s2 = 1; s2 < 16; s2 <<= 1) mx = fmaxf(mx, __shfl_xor(mx, s2, 32));
    float sum = 0.f;
#pragma unroll
    for (int j2 = 0; j2 < 4; ++j2) { lv[j2] = __expf(lv[j2] - mx); sum += lv[j2]; }
#pragma unroll
    for (int s2 = 1; s2 < 16; s2 <<= 1) sum += __shfl_xor(sum, s2, 32);
    float inv = 1.f / sum;
    int prow = r + 8 * hi;
#pragma unroll
    for (int j2 = 0; j2 < 4; ++j2)
      Ps[mt][prow * 72 + j2 * 16 + nl] = (_Float16)(lv[j2] * inv);
  }

  // P(16x64) @ V(64x32)
  v8f o[2] = {vzero, vzero};
#pragma unroll
  for (int ks = 0; ks < 2; ++ks) {
    v16h ap = ld_fragA(Ps[mt], 72, ks * 32);
#pragma unroll
    for (int nt = 0; nt < 2; ++nt) {
      v16h bv = ld_fragB(Vt + nt * 16 * 72, 72, ks * 32);
      o[nt] = wmma_f16(ap, bv, o[nt]);
    }
  }

  // scatter back to token order (implements window-reverse + inverse roll)
#pragma unroll
  for (int nt = 0; nt < 2; ++nt) {
    int c = head * 32 + nt * 16 + nl;
#pragma unroll
    for (int r = 0; r < 8; ++r) {
      int p = mt * 16 + r + 8 * hi;
      if (p < 49) outp[(size_t)tok[p] * DIMC + c] = (_Float16)o[nt][r];
    }
  }
}

// ---------------- Launcher ----------------
extern "C" void kernel_launch(void* const* d_in, const int* in_sizes, int n_in,
                              void* d_out, int out_size, void* d_ws, size_t ws_size,
                              hipStream_t stream) {
  (void)in_sizes; (void)n_in; (void)out_size; (void)ws_size;
  const float* x      = (const float*)d_in[0];
  const float* ln1_s  = (const float*)d_in[1];
  const float* ln1_b  = (const float*)d_in[2];
  const float* qkv_w  = (const float*)d_in[3];
  const float* qkv_b  = (const float*)d_in[4];
  const float* proj_w = (const float*)d_in[5];
  const float* proj_b = (const float*)d_in[6];
  const float* rpb    = (const float*)d_in[7];
  const float* ln2_s  = (const float*)d_in[8];
  const float* ln2_b  = (const float*)d_in[9];
  const float* mlp_w1 = (const float*)d_in[10];
  const float* mlp_b1 = (const float*)d_in[11];
  const float* mlp_w2 = (const float*)d_in[12];
  const float* mlp_b2 = (const float*)d_in[13];
  float* out = (float*)d_out;

  char* wsb = (char*)d_ws;
  _Float16* wqh  = (_Float16*)(wsb + 0);          // qkv_w^T  : [1152][384]
  _Float16* wph  = (_Float16*)(wsb + 884736);     // proj_w^T : [384][384]
  _Float16* w1h  = (_Float16*)(wsb + 1179648);    // mlp_w1^T : [1536][384]
  _Float16* w2h  = (_Float16*)(wsb + 2359296);    // mlp_w2^T : [384][1536]
  _Float16* lnh  = (_Float16*)(wsb + 4194304);    // 100352*384 f16 (LN1, reused for LN2)
  _Float16* qkvh = (_Float16*)(wsb + 81264640);   // 100352*1152 f16
  _Float16* atth = (_Float16*)(wsb + 312475648);  // 100352*384 f16
  _Float16* hidh = (_Float16*)(wsb + 81264640);   // 100352*1536 f16 (reuses qkv+att region)

  cvt_transpose_f16<<<1728, 256, 0, stream>>>(qkv_w, wqh, 384, 1152);
  cvt_transpose_f16<<<576,  256, 0, stream>>>(proj_w, wph, 384, 384);
  cvt_transpose_f16<<<2304, 256, 0, stream>>>(mlp_w1, w1h, 384, 1536);
  cvt_transpose_f16<<<2304, 256, 0, stream>>>(mlp_w2, w2h, 1536, 384);

  ln_f16_kernel<<<TOKENS / 8, 256, 0, stream>>>(x, ln1_s, ln1_b, lnh, TOKENS);
  gemm_f16_kernel<0><<<dim3(9, 784), 256, 0, stream>>>(lnh, wqh, qkv_b, nullptr, qkvh,
                                                       TOKENS, 1152, 384);
  attn_kernel<<<dim3(12, 2048), 128, 0, stream>>>(qkvh, rpb, atth);
  gemm_f16_kernel<1><<<dim3(3, 784), 256, 0, stream>>>(atth, wph, proj_b, x, out,
                                                       TOKENS, 384, 384);
  ln_f16_kernel<<<TOKENS / 8, 256, 0, stream>>>(out, ln2_s, ln2_b, lnh, TOKENS);
  gemm_f16_kernel<2><<<dim3(12, 784), 256, 0, stream>>>(lnh, w1h, mlp_b1, nullptr, hidh,
                                                        TOKENS, 1536, 384);
  gemm_f16_kernel<1><<<dim3(3, 784), 256, 0, stream>>>(hidh, w2h, mlp_b2, out, out,
                                                       TOKENS, 384, 1536);
}
